// Indexer_16466904613592
// MI455X (gfx1250) — matrix operand order repaired and checked by reading, not compiled, not run on hardware
//
#include <hip/hip_runtime.h>
#include <hip/hip_bf16.h>

typedef __attribute__((ext_vector_type(16))) _Float16 v16h;
typedef __attribute__((ext_vector_type(8)))  _Float16 v8h;
typedef __attribute__((ext_vector_type(8)))  float    v8f;

constexpr int Bb   = 2;
constexpr int Tt   = 2048;
constexpr int DIN  = 2048;
constexpr int DKV  = 512;
constexpr int Hh   = 16;
constexpr int HD   = 128;
constexpr int TOPK = 512;

// ---------------------------------------------------------------------------
// Bulk f32 -> f16 conversion (one pass; bandwidth-trivial at 23.3 TB/s).
// n must be a multiple of 8.
// ---------------------------------------------------------------------------
__global__ __launch_bounds__(256)
void cvt_f32_f16(const float* __restrict__ in, _Float16* __restrict__ out, long n) {
  const long i = ((long)blockIdx.x * 256 + threadIdx.x) * 8;
  if (i >= n) return;
  const float4* p = (const float4*)(in + i);
  const float4 a = p[0], b = p[1];
  v8h o;
  o[0] = (_Float16)a.x; o[1] = (_Float16)a.y; o[2] = (_Float16)a.z; o[3] = (_Float16)a.w;
  o[4] = (_Float16)b.x; o[5] = (_Float16)b.y; o[6] = (_Float16)b.z; o[7] = (_Float16)b.w;
  *(v8h*)(out + i) = o;
}

// ---------------------------------------------------------------------------
// WMMA GEMM:  C[M,N] = A[M,K] * Bt[N,K]^T + bias[N]   (f16 in, f32 out).
// Block tile 128x128, 8 waves (4 row-bands x 2 col-bands); each wave owns
// two 16-row A bands x four 16-col B bands = 8 WMMA per 32-wide K step.
// All fragments are loaded before the WMMA burst so the matrix pipe issues
// back-to-back.  Staging via CDNA5 async-to-LDS (ASYNCcnt), double buffered.
// M % 128 == 0, N % 128 == 0, K % 32 == 0.
// ---------------------------------------------------------------------------
__global__ __launch_bounds__(256)
void gemm_wmma_async(const _Float16* __restrict__ A, const _Float16* __restrict__ Bt,
                     const float* __restrict__ bias, float* __restrict__ C,
                     int M, int N, int K) {
  __shared__ _Float16 lA[2][128 * 32];  // 16 KB
  __shared__ _Float16 lB[2][128 * 32];  // 16 KB

  const int tid  = threadIdx.x;
  const int lane = tid & 31;
  const int wave = tid >> 5;
  const int wm   = wave & 3;   // row band (16 rows, x2 via +64)
  const int wn   = wave >> 2;  // col band (64 cols)
  const int row0 = blockIdx.y * 128;
  const int col0 = blockIdx.x * 128;
  const int half = lane >> 4;
  const int n15  = lane & 15;

  // staging: A and B each 128x32 f16 = 512 b128 xfers -> 2 per thread each
  const int srow = tid >> 1, sseg = (tid & 1) * 16;

  v8f acc[8] = {{}, {}, {}, {}, {}, {}, {}, {}};

  auto issue = [&](int buf, int kk) {
    const unsigned la = (unsigned)(unsigned long long)&lA[buf][srow * 32 + sseg];
    const unsigned ga = (unsigned)(((size_t)(row0 + srow) * K + kk + sseg) * sizeof(_Float16));
    asm volatile("global_load_async_to_lds_b128 %0, %1, %2"
                 :: "v"(la), "v"(ga), "s"(A) : "memory");
    asm volatile("global_load_async_to_lds_b128 %0, %1, %2"
                 :: "v"(la + 16u), "v"(ga + 16u), "s"(A) : "memory");
    const unsigned lb = (unsigned)(unsigned long long)&lB[buf][srow * 32 + sseg];
    const unsigned gb = (unsigned)(((size_t)(col0 + srow) * K + kk + sseg) * sizeof(_Float16));
    asm volatile("global_load_async_to_lds_b128 %0, %1, %2"
                 :: "v"(lb), "v"(gb), "s"(Bt) : "memory");
    asm volatile("global_load_async_to_lds_b128 %0, %1, %2"
                 :: "v"(lb + 16u), "v"(gb + 16u), "s"(Bt) : "memory");
  };

  const int nsteps = K / 32;
  issue(0, 0);
  for (int i = 0; i < nsteps; ++i) {
    const int buf = i & 1;
    asm volatile("s_wait_asynccnt 0x0" ::: "memory");
    __syncthreads();                       // this buffer ready; prev reads done
    if (i + 1 < nsteps) issue(buf ^ 1, (i + 1) * 32);

    // ---- load ALL fragments first ----
    v16h af0, af1, bf0, bf1, bf2, bf3;
#pragma unroll
    for (int q = 0; q < 16; ++q) {
      const int ka = (half << 4) + q;                          // A frag K index
      af0[q] = lA[buf][(wm * 16 + n15) * 32 + ka];
      af1[q] = lA[buf][(64 + wm * 16 + n15) * 32 + ka];
      const int kb = (half << 4) + q;                          // B frag K index
      bf0[q] = lB[buf][(wn * 64 +  0 + n15) * 32 + kb];
      bf1[q] = lB[buf][(wn * 64 + 16 + n15) * 32 + kb];
      bf2[q] = lB[buf][(wn * 64 + 32 + n15) * 32 + kb];
      bf3[q] = lB[buf][(wn * 64 + 48 + n15) * 32 + kb];
    }
    // ---- 8 WMMAs back-to-back ----
    acc[0] = __builtin_amdgcn_wmma_f32_16x16x32_f16(false, af0, false, bf0, (short)0, acc[0], false, false);
    acc[1] = __builtin_amdgcn_wmma_f32_16x16x32_f16(false, af0, false, bf1, (short)0, acc[1], false, false);
    acc[2] = __builtin_amdgcn_wmma_f32_16x16x32_f16(false, af0, false, bf2, (short)0, acc[2], false, false);
    acc[3] = __builtin_amdgcn_wmma_f32_16x16x32_f16(false, af0, false, bf3, (short)0, acc[3], false, false);
    acc[4] = __builtin_amdgcn_wmma_f32_16x16x32_f16(false, af1, false, bf0, (short)0, acc[4], false, false);
    acc[5] = __builtin_amdgcn_wmma_f32_16x16x32_f16(false, af1, false, bf1, (short)0, acc[5], false, false);
    acc[6] = __builtin_amdgcn_wmma_f32_16x16x32_f16(false, af1, false, bf2, (short)0, acc[6], false, false);
    acc[7] = __builtin_amdgcn_wmma_f32_16x16x32_f16(false, af1, false, bf3, (short)0, acc[7], false, false);
  }

#pragma unroll
  for (int a2 = 0; a2 < 2; ++a2)
#pragma unroll
    for (int s = 0; s < 4; ++s)
#pragma unroll
      for (int r = 0; r < 8; ++r) {
        const int m = row0 + a2 * 64 + wm * 16 + half * 8 + r;
        const int c = col0 + wn * 64 + s * 16 + n15;
        C[(size_t)m * N + c] = acc[a2 * 4 + s][r] + bias[c];
      }
}

// ---------------------------------------------------------------------------
// LayerNorm + RoPE for q, writing f16 in WMMA A-fragment order:
// qfrag[((b*128 + t/16)*16 + h)*4 + c][lane*16 + i]
// ---------------------------------------------------------------------------
__global__ __launch_bounds__(256)
void ln_rope_q(const float* __restrict__ qraw, const float* __restrict__ g,
               const float* __restrict__ be, _Float16* __restrict__ qfrag) {
  const int bt = blockIdx.x;
  const int b = bt / Tt, t = bt % Tt;
  const float* row = qraw + (size_t)bt * (Hh * HD);
  __shared__ float red[256];

  const int j0 = threadIdx.x * 8;
  float vals[8];
  float s = 0.f;
#pragma unroll
  for (int u = 0; u < 8; ++u) { vals[u] = row[j0 + u]; s += vals[u]; }
  red[threadIdx.x] = s; __syncthreads();
  for (int off = 128; off > 0; off >>= 1) {
    if (threadIdx.x < off) red[threadIdx.x] += red[threadIdx.x + off];
    __syncthreads();
  }
  const float mean = red[0] * (1.0f / 2048.0f);
  __syncthreads();
  s = 0.f;
#pragma unroll
  for (int u = 0; u < 8; ++u) { float d = vals[u] - mean; s += d * d; }
  red[threadIdx.x] = s; __syncthreads();
  for (int off = 128; off > 0; off >>= 1) {
    if (threadIdx.x < off) red[threadIdx.x] += red[threadIdx.x + off];
    __syncthreads();
  }
  const float rstd = rsqrtf(red[0] * (1.0f / 2048.0f) + 1e-5f);

#pragma unroll
  for (int u = 0; u < 8; ++u)
    vals[u] = (vals[u] - mean) * rstd * g[j0 + u] + be[j0 + u];

  if (j0 >= Hh * 64) {  // rope half: interleaved pair rotation
#pragma unroll
    for (int p = 0; p < 4; ++p) {
      const int rr = ((j0 + 2 * p) - Hh * 64) & 63;
      const int ph = rr >> 1;
      const float inv = powf(10000.0f, -(float)ph / 32.0f);
      const float ang = (float)t * inv;
      const float cs = cosf(ang), sn = sinf(ang);
      const float x0 = vals[2 * p], x1 = vals[2 * p + 1];
      vals[2 * p]     = x0 * cs - x1 * sn;
      vals[2 * p + 1] = x0 * sn + x1 * cs;
    }
  }

  const int ttile = t >> 4, m = t & 15;
#pragma unroll
  for (int u = 0; u < 8; ++u) {
    const int j = j0 + u;
    int h, d;
    if (j < Hh * 64) { h = j >> 6; d = j & 63; }
    else             { int j2 = j - Hh * 64; h = j2 >> 6; d = 64 + (j2 & 63); }
    const int c = d >> 5, w32 = d & 31;
    const int gg = w32 >> 4, rem = w32 & 15, hf = rem >> 3, wb = rem & 7;
    const int i = gg * 8 + wb;
    const int ln = hf * 16 + m;
    const size_t fb = (((size_t)(b * (Tt / 16) + ttile) * Hh + h) * 4 + c);
    qfrag[fb * 512 + ln * 16 + i] = (_Float16)vals[u];
  }
}

// ---------------------------------------------------------------------------
// LayerNorm + RoPE for k: writes f32 k_cache and f16 WMMA B-fragment order:
// kfrag[(b*128 + k/16)*4 + c][lane*16 + i]
// ---------------------------------------------------------------------------
__global__ __launch_bounds__(128)
void ln_rope_k(const float* __restrict__ kraw, const float* __restrict__ g,
               const float* __restrict__ be, float* __restrict__ kcache,
               _Float16* __restrict__ kfrag) {
  const int bk = blockIdx.x;
  const int b = bk / Tt, k = bk % Tt;
  const float* row = kraw + (size_t)bk * HD;
  __shared__ float red[128];
  __shared__ float nrm[128];

  const int d = threadIdx.x;
  const float v = row[d];
  red[d] = v; __syncthreads();
  for (int off = 64; off > 0; off >>= 1) {
    if (d < off) red[d] += red[d + off];
    __syncthreads();
  }
  const float mean = red[0] * (1.0f / 128.0f);
  __syncthreads();
  const float dv = v - mean;
  red[d] = dv * dv; __syncthreads();
  for (int off = 64; off > 0; off >>= 1) {
    if (d < off) red[d] += red[d + off];
    __syncthreads();
  }
  const float rstd = rsqrtf(red[0] * (1.0f / 128.0f) + 1e-5f);

  const float x = dv * rstd * g[d] + be[d];
  nrm[d] = x; __syncthreads();

  float outv;
  if (d < 64) {
    outv = x;
  } else {
    const int rr = d - 64, ph = rr >> 1;
    const float inv = powf(10000.0f, -(float)ph / 32.0f);
    const float ang = (float)k * inv;
    const float cs = cosf(ang), sn = sinf(ang);
    const float x0 = nrm[64 + 2 * ph], x1 = nrm[64 + 2 * ph + 1];
    outv = ((rr & 1) == 0) ? (x0 * cs - x1 * sn) : (x0 * sn + x1 * cs);
  }
  kcache[(size_t)bk * HD + d] = outv;

  const int ktile = k >> 4, n = k & 15;
  const int c = d >> 5, w32 = d & 31, hf = (w32 >> 3) & 1, i = ((w32 >> 4) << 3) + (w32 & 7);
  const int ln = hf * 16 + n;
  const size_t fb = ((size_t)(b * (Tt / 16) + ktile)) * 4 + c;
  kfrag[fb * 512 + ln * 16 + i] = (_Float16)outv;
}

// ---------------------------------------------------------------------------
// w projection (N=16) + LayerNorm over 16 — VALU only, tiny cost.
// ---------------------------------------------------------------------------
__global__ __launch_bounds__(256)
void proj_ln_w(const float* __restrict__ x, const float* __restrict__ Ww,
               const float* __restrict__ wb, const float* __restrict__ g,
               const float* __restrict__ be, float* __restrict__ widx) {
  const int bt = blockIdx.x;
  const float* row = x + (size_t)bt * DIN;
  const int h = threadIdx.x >> 4, sub = threadIdx.x & 15;
  float s = 0.f;
  for (int i = sub; i < DIN; i += 16) s += row[i] * Ww[(size_t)h * DIN + i];
  __shared__ float part[16][16];
  __shared__ float raw[16];
  part[h][sub] = s; __syncthreads();
  if (sub == 0) {
    float t = 0.f;
#pragma unroll
    for (int i = 0; i < 16; ++i) t += part[h][i];
    raw[h] = t + wb[h];
  }
  __syncthreads();
  if (threadIdx.x == 0) {
    float m = 0.f;
#pragma unroll
    for (int i = 0; i < 16; ++i) m += raw[i];
    m *= (1.0f / 16.0f);
    float v = 0.f;
#pragma unroll
    for (int i = 0; i < 16; ++i) { float dd = raw[i] - m; v += dd * dd; }
    const float rs = rsqrtf(v * (1.0f / 16.0f) + 1e-5f);
#pragma unroll
    for (int i = 0; i < 16; ++i)
      widx[(size_t)bt * 16 + i] = (raw[i] - m) * rs * g[i] + be[i];
  }
}

// ---------------------------------------------------------------------------
// Score kernel: idx_scores[b,t,k] = sum_h w[b,t,h] * relu(q_full . k_full)
// + causal mask.  One block = 16 t-rows x 128 k-cols; each wave one 16x16
// tile.  k fragments hoisted out of the head loop (invariant).
// ---------------------------------------------------------------------------
__global__ __launch_bounds__(256)
void score_wmma(const _Float16* __restrict__ qfrag,
                const _Float16* __restrict__ kfrag,
                const float* __restrict__ widx, float* __restrict__ scores) {
  const int b  = blockIdx.z;
  const int t0 = blockIdx.y * 16;
  const int kb = blockIdx.x * 128;
  const int tid = threadIdx.x, lane = tid & 31, wave = tid >> 5;
  const int ktile = (kb >> 4) + wave;
  const int half = lane >> 4, n = lane & 15;

  __shared__ float lw[16][16];
  {
    const int r = tid >> 4, h = tid & 15;
    lw[r][h] = widx[((size_t)b * Tt + t0 + r) * 16 + h];
  }
  __syncthreads();

  const v16h* qv = (const v16h*)qfrag;
  const v16h* kv = (const v16h*)kfrag;
  const size_t qbase = ((size_t)(b * (Tt / 16) + (t0 >> 4)) * Hh) * 4;
  const size_t kbase = ((size_t)(b * (Tt / 16) + ktile)) * 4;

  v16h kf[4];
#pragma unroll
  for (int c = 0; c < 4; ++c) kf[c] = kv[(kbase + c) * 32 + lane];

  float out[8];
#pragma unroll
  for (int r = 0; r < 8; ++r) out[r] = 0.f;

  for (int h = 0; h < Hh; ++h) {
    v8f acc = {};
#pragma unroll
    for (int c = 0; c < 4; ++c) {
      v16h a = qv[(qbase + (size_t)h * 4 + c) * 32 + lane];
      acc = __builtin_amdgcn_wmma_f32_16x16x32_f16(false, a, false, kf[c],
                                                   (short)0, acc, false, false);
    }
#pragma unroll
    for (int r = 0; r < 8; ++r) {
      float sv = acc[r];
      sv = sv > 0.f ? sv : 0.f;
      out[r] += lw[half * 8 + r][h] * sv;
    }
  }

#pragma unroll
  for (int r = 0; r < 8; ++r) {
    const int t = t0 + half * 8 + r;
    const int kk = (ktile << 4) + n;
    const float val = (kk <= t) ? out[r] : -1e9f;
    scores[((size_t)b * Tt + t) * Tt + kk] = val;
  }
}

// ---------------------------------------------------------------------------
// Top-512 of 2048 per row: bitonic sort (ascending on negated score) in LDS.
// ---------------------------------------------------------------------------
__global__ __launch_bounds__(256)
void topk_bitonic(const float* __restrict__ scores, int* __restrict__ topi) {
  const int bt = blockIdx.x;
  __shared__ float ks[2048];
  __shared__ int   vs[2048];
  const float* row = scores + (size_t)bt * Tt;
  for (int j = threadIdx.x; j < 2048; j += 256) { ks[j] = -row[j]; vs[j] = j; }
  __syncthreads();

  for (unsigned k = 2; k <= 2048; k <<= 1) {
    for (unsigned j = k >> 1; j > 0; j >>= 1) {
      for (unsigned i = threadIdx.x; i < 2048; i += 256) {
        const unsigned ixj = i ^ j;
        if (ixj > i) {
          const bool up = ((i & k) == 0);
          const float a = ks[i], c = ks[ixj];
          const bool sw = up ? (a > c) : (a < c);
          if (sw) {
            ks[i] = c; ks[ixj] = a;
            const int tv = vs[i]; vs[i] = vs[ixj]; vs[ixj] = tv;
          }
        }
      }
      __syncthreads();
    }
  }
  for (int j = threadIdx.x; j < TOPK; j += 256)
    topi[(size_t)bt * TOPK + j] = vs[j];
}

// ---------------------------------------------------------------------------
extern "C" void kernel_launch(void* const* d_in, const int* in_sizes, int n_in,
                              void* d_out, int out_size, void* d_ws, size_t ws_size,
                              hipStream_t stream) {
  (void)in_sizes; (void)n_in; (void)out_size; (void)ws_size;
  const float* x    = (const float*)d_in[0];
  const float* kvm  = (const float*)d_in[1];
  // d_in[2] = mask (causal tril) — recomputed analytically in score_wmma
  const float* Wq_w = (const float*)d_in[3];
  const float* Wq_b = (const float*)d_in[4];
  const float* Wk_w = (const float*)d_in[5];
  const float* Wk_b = (const float*)d_in[6];
  const float* Ww_w = (const float*)d_in[7];
  const float* Ww_b = (const float*)d_in[8];
  const float* qn_g = (const float*)d_in[9];
  const float* qn_b = (const float*)d_in[10];
  const float* kn_g = (const float*)d_in[11];
  const float* kn_b = (const float*)d_in[12];
  const float* wn_g = (const float*)d_in[13];
  const float* wn_b = (const float*)d_in[14];

  // ---- output regions (tuple order: top_indices, idx_scores, k_cache) ----
  int*   topi   = (int*)d_out;
  float* scores = (float*)d_out + (size_t)Bb * Tt * TOPK;
  float* kcache = scores + (size_t)Bb * Tt * Tt;

  // ---- workspace carve (~84 MB) ----
  char* p = (char*)d_ws;
  float* qraw = (float*)p;        p += (size_t)Bb * Tt * (Hh * HD) * sizeof(float);
  float* kraw = (float*)p;        p += (size_t)Bb * Tt * HD * sizeof(float);
  _Float16* qfrag = (_Float16*)p; p += (size_t)Bb * Tt * (Hh * HD) * sizeof(_Float16);
  _Float16* kfrag = (_Float16*)p; p += (size_t)Bb * Tt * HD * sizeof(_Float16);
  float* widx = (float*)p;        p += (size_t)Bb * Tt * Hh * sizeof(float);
  _Float16* xh  = (_Float16*)p;   p += (size_t)Bb * Tt * DIN * sizeof(_Float16);
  _Float16* kvh = (_Float16*)p;   p += (size_t)Bb * Tt * DKV * sizeof(_Float16);
  _Float16* Wqh = (_Float16*)p;   p += (size_t)(Hh * HD) * DIN * sizeof(_Float16);
  _Float16* Wkh = (_Float16*)p;   p += (size_t)HD * DKV * sizeof(_Float16);

  const int M = Bb * Tt;  // 4096

  // f32 -> f16 one-pass conversions
  {
    const long nx = (long)M * DIN, nkv = (long)M * DKV;
    const long nwq = (long)(Hh * HD) * DIN, nwk = (long)HD * DKV;
    cvt_f32_f16<<<(unsigned)(nx  / (8 * 256)), 256, 0, stream>>>(x,    xh,  nx);
    cvt_f32_f16<<<(unsigned)(nkv / (8 * 256)), 256, 0, stream>>>(kvm,  kvh, nkv);
    cvt_f32_f16<<<(unsigned)(nwq / (8 * 256)), 256, 0, stream>>>(Wq_w, Wqh, nwq);
    cvt_f32_f16<<<(unsigned)(nwk / (8 * 256)), 256, 0, stream>>>(Wk_w, Wkh, nwk);
  }

  // q projection: (4096 x 2048) x (2048 x 2048)^T
  gemm_wmma_async<<<dim3((Hh * HD) / 128, M / 128), 256, 0, stream>>>(
      xh, Wqh, Wq_b, qraw, M, Hh * HD, DIN);
  // k projection: (4096 x 512) x (128 x 512)^T
  gemm_wmma_async<<<dim3(HD / 128, M / 128), 256, 0, stream>>>(
      kvh, Wkh, Wk_b, kraw, M, HD, DKV);
  // w projection + LN
  proj_ln_w<<<M, 256, 0, stream>>>(x, Ww_w, Ww_b, wn_g, wn_b, widx);
  // LN + RoPE + fragment swizzle
  ln_rope_q<<<M, 256, 0, stream>>>(qraw, qn_g, qn_b, qfrag);
  ln_rope_k<<<M, 128, 0, stream>>>(kraw, kn_g, kn_b, kcache, kfrag);
  // scores (WMMA) + mask
  score_wmma<<<dim3(Tt / 128, Tt / 16, Bb), 256, 0, stream>>>(qfrag, kfrag, widx, scores);
  // top-k
  topk_bitonic<<<M, 256, 0, stream>>>(scores, topi);
}
